// CausalSelfAttention_29703993819951
// MI455X (gfx1250) — 16-bit and 8-bit matrix operand order repaired and checked by reading, compiled, not run on hardware
//
#include <hip/hip_runtime.h>
#include <math.h>

typedef __attribute__((ext_vector_type(16))) _Float16 v16h;
typedef __attribute__((ext_vector_type(8)))  _Float16 h8;
typedef __attribute__((ext_vector_type(8)))  float    v8f;
typedef __attribute__((ext_vector_type(8)))  int      v8i;
typedef __attribute__((ext_vector_type(2)))  _Float16 h2;

#define DIM        1024
#define NUM_HEADS  16
#define NUM_KVH    4
#define HEAD_DIM   64
#define BSZ        8
#define SEQ        1024
#define ROWS       (BSZ*SEQ)
#define KVDIM      (NUM_KVH*HEAD_DIM)

#define PSTR 34
#define KSTR 72
#define VSTR 34

__global__ void alpha_kernel(const float* __restrict__ W, int n, float* alpha) {
    __shared__ float red[256];
    float s = 0.f;
    for (int i = threadIdx.x; i < n; i += 256) s += fabsf(W[i]);
    red[threadIdx.x] = s;
    __syncthreads();
    for (int st = 128; st > 0; st >>= 1) {
        if (threadIdx.x < st) red[threadIdx.x] += red[threadIdx.x + st];
        __syncthreads();
    }
    if (threadIdx.x == 0) alpha[0] = fmaxf(red[0] / (float)n, 1e-8f);
}

__global__ void quant_w_kernel(const float* __restrict__ W, const float* alpha,
                               signed char* __restrict__ Wt, int n) {
    int i = blockIdx.x * 256 + threadIdx.x;
    if (i < n) {
        float a = alpha[0];
        Wt[i] = (signed char)rintf(fminf(fmaxf(W[i] / a, -1.f), 1.f));
    }
}

__global__ void quant_rows_kernel(const float* __restrict__ X,
                                  signed char* __restrict__ Xq,
                                  float* __restrict__ gamma, int K) {
    __shared__ float red[256];
    int row = blockIdx.x, tid = threadIdx.x;
    const float* xr = X + (size_t)row * K;
    float mx = 0.f;
    for (int i = tid; i < K; i += 256) mx = fmaxf(mx, fabsf(xr[i]));
    red[tid] = mx;
    __syncthreads();
    for (int st = 128; st > 0; st >>= 1) {
        if (tid < st) red[tid] = fmaxf(red[tid], red[tid + st]);
        __syncthreads();
    }
    float g = fmaxf(red[0], 1e-8f) / 127.0f;
    if (tid == 0) gamma[row] = g;
    float inv = 1.0f / g;
    for (int i = tid; i < K; i += 256) {
        float q = rintf(fminf(fmaxf(xr[i] * inv, -128.f), 127.f));
        Xq[(size_t)row * K + i] = (signed char)q;
    }
}

__global__ void gemm_iu8_kernel(const signed char* __restrict__ Xq,
                                const signed char* __restrict__ Wt,
                                const float* __restrict__ gamma,
                                const float* __restrict__ alpha,
                                float* __restrict__ Out,
                                int M, int N, int K) {
    int w = threadIdx.x >> 5, lane = threadIdx.x & 31;
    int t = blockIdx.x * 8 + w;
    int tilesM = M >> 4;
    int tm = t % tilesM, tn4 = t / tilesM;
    if (tn4 >= (N >> 6)) return;
    int n = lane & 15, hi = lane >> 4;
    const signed char* ap = Xq + (size_t)(tm * 16 + n) * K;
    const signed char* bp = Wt + (size_t)(tn4 * 64 + n) * K;

    auto loadA = [&](int kc, v8i& A) {
#pragma unroll
        for (int j = 0; j < 8; ++j) {
            int ka = kc + ((j >> 2) << 5) + (hi << 4) + ((j & 3) << 2);
            A[j] = *reinterpret_cast<const int*>(ap + ka);
        }
    };
    auto loadB = [&](int kc, v8i* B) {
#pragma unroll
        for (int u = 0; u < 4; ++u)
#pragma unroll
            for (int j = 0; j < 8; ++j) {
                int kb = kc + ((j >> 2) << 5) + (hi << 4) + ((j & 3) << 2);
                B[u][j] = *reinterpret_cast<const int*>(bp + (size_t)u * 16 * K + kb);
            }
    };

    v8i acc[4] = {};
    v8i a0, b0[4], a1, b1[4];
    loadA(0, a0);
    loadB(0, b0);
    for (int kc = 0; kc < K; kc += 128) {
        if (kc + 64 < K) {
            __builtin_prefetch(ap + kc + 128, 0, 3);
            loadA(kc + 64, a1);
            loadB(kc + 64, b1);
        }
#pragma unroll
        for (int u = 0; u < 4; ++u)
            acc[u] = __builtin_amdgcn_wmma_i32_16x16x64_iu8(true, a0, true, b0[u],
                                                            acc[u], false, false);
        if (kc + 128 < K) {
            __builtin_prefetch(ap + kc + 192, 0, 3);
            loadA(kc + 128, a0);
            loadB(kc + 128, b0);
        }
        if (kc + 64 < K) {
#pragma unroll
            for (int u = 0; u < 4; ++u)
                acc[u] = __builtin_amdgcn_wmma_i32_16x16x64_iu8(true, a1, true, b1[u],
                                                                acc[u], false, false);
        }
    }
    float al = alpha[0];
#pragma unroll
    for (int u = 0; u < 4; ++u)
#pragma unroll
        for (int r = 0; r < 8; ++r) {
            int m = tm * 16 + r + 8 * hi;
            Out[(size_t)m * N + tn4 * 64 + u * 16 + n] =
                (float)acc[u][r] * gamma[m] * al;
        }
}

__global__ void normrope_kernel(const float* __restrict__ In,
                                _Float16* __restrict__ Out,
                                const float* __restrict__ gain,
                                int nh, int mode, float scale, int totalWaves) {
    int gid = blockIdx.x * blockDim.x + threadIdx.x;
    int wid = gid >> 5, lane = gid & 31;
    if (wid >= totalWaves) return;
    int h = wid % nh, row = wid / nh;
    int pos = row % SEQ, b = row / SEQ;
    const float* src = In + (size_t)row * (nh * HEAD_DIM) + h * HEAD_DIM;
    float x1 = src[lane], x2 = src[lane + 32];
    float s = scale;
    if (mode >= 1) {
        float ss = x1 * x1 + x2 * x2;
#pragma unroll
        for (int m = 1; m < 32; m <<= 1) ss += __shfl_xor(ss, m, 32);
        float r = rsqrtf(ss * (1.0f / 64.0f) + 1.1920929e-7f);
        x1 *= r; x2 *= r;
        float invf = __powf(10000.0f, -(float)lane * (1.0f / 32.0f));
        float ang = (float)pos * invf, sn, c;
        __sincosf(ang, &sn, &c);
        float o1 =  x1 * c + x2 * sn;
        float o2 = -x1 * sn + x2 * c;
        x1 = o1; x2 = o2;
        if (mode == 2) s *= gain[h];
    }
    _Float16* dst = Out + ((size_t)(b * nh + h) * SEQ + pos) * HEAD_DIM;
    dst[lane]      = (_Float16)(x1 * s);
    dst[lane + 32] = (_Float16)(x2 * s);
}

__global__ void attn_kernel(const _Float16* __restrict__ Q,
                            const _Float16* __restrict__ Kh,
                            const _Float16* __restrict__ Vh,
                            float* __restrict__ Y) {
    __shared__ _Float16 ldsP[8 * 16 * PSTR];
    __shared__ _Float16 ldsK[32 * KSTR];
    __shared__ _Float16 ldsVT[64 * VSTR];
    int tid = threadIdx.x;
    int w = tid >> 5, lane = tid & 31;
    int n = lane & 15, hi = lane >> 4;
    int bh = blockIdx.x >> 3, blk = blockIdx.x & 7;
    int b = bh >> 4, h = bh & 15;
    int bkv = b * NUM_KVH + (h >> 2);
    int q0 = blk * 128;
    int qtile = q0 + w * 16;
    const _Float16* Qb = Q  + (size_t)bh  * SEQ * HEAD_DIM;
    const _Float16* Kb = Kh + (size_t)bkv * SEQ * HEAD_DIM;
    const _Float16* Vb = Vh + (size_t)bkv * SEQ * HEAD_DIM;
    _Float16* myP = &ldsP[w * 16 * PSTR];

    int skv = tid >> 3;
    int sd  = (tid & 7) << 3;

    int qrow = qtile + n;
    v16h qa0, qa1;
#pragma unroll
    for (int j = 0; j < 8; ++j) {
        int d = ((j >> 2) << 4) + (hi << 3) + ((j & 3) << 1);
        h2 p0 = *reinterpret_cast<const h2*>(Qb + (size_t)qrow * 64 + d);
        h2 p1 = *reinterpret_cast<const h2*>(Qb + (size_t)qrow * 64 + 32 + d);
        qa0[2*j] = p0[0]; qa0[2*j+1] = p0[1];
        qa1[2*j] = p1[0]; qa1[2*j+1] = p1[1];
    }

    v8f Oc[4] = {};
    float Mrow[8], Lrow[8];
#pragma unroll
    for (int r = 0; r < 8; ++r) { Mrow[r] = -INFINITY; Lrow[r] = 0.f; }

    int nch = (q0 >> 5) + 4;
    for (int c = 0; c < nch; ++c) {
        int cbase = c << 5;
        __syncthreads();
        h8 kv8 = *reinterpret_cast<const h8*>(Kb + (size_t)(cbase + skv) * 64 + sd);
        h8 vv8 = *reinterpret_cast<const h8*>(Vb + (size_t)(cbase + skv) * 64 + sd);
        *reinterpret_cast<h8*>(&ldsK[skv * KSTR + sd]) = kv8;
#pragma unroll
        for (int e = 0; e < 8; ++e) ldsVT[(sd + e) * VSTR + skv] = vv8[e];
        __syncthreads();

        bool active = cbase <= qtile + 15;
        float corr[8];
        if (active) {
            v8f sc[2];
#pragma unroll
            for (int t = 0; t < 2; ++t) {
                const _Float16* kr = &ldsK[(t * 16 + n) * KSTR];
                v16h kb0, kb1;
#pragma unroll
                for (int j = 0; j < 8; ++j) {
                    int d = ((j >> 2) << 4) + (hi << 3) + ((j & 3) << 1);
                    h2 p0 = *reinterpret_cast<const h2*>(kr + d);
                    h2 p1 = *reinterpret_cast<const h2*>(kr + 32 + d);
                    kb0[2*j] = p0[0]; kb0[2*j+1] = p0[1];
                    kb1[2*j] = p1[0]; kb1[2*j+1] = p1[1];
                }
                v8f z = {};
                z = __builtin_amdgcn_wmma_f32_16x16x32_f16(false, qa0, false, kb0,
                                                           (short)0, z, false, false);
                z = __builtin_amdgcn_wmma_f32_16x16x32_f16(false, qa1, false, kb1,
                                                           (short)0, z, false, false);
                sc[t] = z;
            }
#pragma unroll
            for (int r = 0; r < 8; ++r) {
                int qr = qtile + r + 8 * hi;
                float v0 = (cbase + n      <= qr) ? sc[0][r] : -3.0e38f;
                float v1 = (cbase + 16 + n <= qr) ? sc[1][r] : -3.0e38f;
                float mx = fmaxf(v0, v1);
#pragma unroll
                for (int m = 1; m < 16; m <<= 1) mx = fmaxf(mx, __shfl_xor(mx, m, 32));
                float nm = fmaxf(Mrow[r], mx);
                float cr = __expf(Mrow[r] - nm);
                float p0 = __expf(v0 - nm), p1 = __expf(v1 - nm);
                float rs = p0 + p1;
#pragma unroll
                for (int m = 1; m < 16; m <<= 1) rs += __shfl_xor(rs, m, 32);
                Lrow[r] = Lrow[r] * cr + rs;
                Mrow[r] = nm;
                corr[r] = cr;
                myP[(r + 8 * hi) * PSTR + n]      = (_Float16)p0;
                myP[(r + 8 * hi) * PSTR + 16 + n] = (_Float16)p1;
            }
#pragma unroll
            for (int cc = 0; cc < 4; ++cc)
#pragma unroll
                for (int r = 0; r < 8; ++r) Oc[cc][r] *= corr[r];
        }
        __syncthreads();
        if (active) {
            v16h pa;
#pragma unroll
            for (int j = 0; j < 8; ++j) {
                int k = ((j >> 2) << 4) + (hi << 3) + ((j & 3) << 1);
                h2 pp = *reinterpret_cast<const h2*>(myP + n * PSTR + k);
                pa[2*j] = pp[0]; pa[2*j+1] = pp[1];
            }
#pragma unroll
            for (int cc = 0; cc < 4; ++cc) {
                const _Float16* vr = &ldsVT[((cc << 4) + n) * VSTR];
                v16h vb;
#pragma unroll
                for (int j = 0; j < 8; ++j) {
                    h2 vp = *reinterpret_cast<const h2*>(vr + ((j >> 2) << 4) + (hi << 3) + ((j & 3) << 1));
                    vb[2*j] = vp[0]; vb[2*j+1] = vp[1];
                }
                Oc[cc] = __builtin_amdgcn_wmma_f32_16x16x32_f16(false, pa, false, vb,
                                                                (short)0, Oc[cc],
                                                                false, false);
            }
        }
    }
#pragma unroll
    for (int cc = 0; cc < 4; ++cc)
#pragma unroll
        for (int r = 0; r < 8; ++r) {
            int qr = qtile + r + 8 * hi;
            Y[((size_t)b * SEQ + qr) * DIM + h * HEAD_DIM + cc * 16 + n] =
                Oc[cc][r] / Lrow[r];
        }
}

extern "C" void kernel_launch(void* const* d_in, const int* in_sizes, int n_in,
                              void* d_out, int out_size, void* d_ws, size_t ws_size,
                              hipStream_t stream) {
    (void)in_sizes; (void)n_in; (void)out_size; (void)ws_size;
    const float* x    = (const float*)d_in[0];
    const float* Wq   = (const float*)d_in[1];
    const float* Wk   = (const float*)d_in[2];
    const float* Wv   = (const float*)d_in[3];
    const float* Wp   = (const float*)d_in[4];
    const float* gain = (const float*)d_in[5];
    float* out = (float*)d_out;

    char* ws = (char*)d_ws;
    size_t off = 0;
    auto alloc = [&](size_t bytes) -> void* {
        void* p = ws + off;
        off = (off + bytes + 255) & ~(size_t)255;
        return p;
    };
    float*       alphas = (float*)alloc(4 * sizeof(float));
    signed char* WqT    = (signed char*)alloc((size_t)DIM * DIM);
    signed char* WkT    = (signed char*)alloc((size_t)KVDIM * DIM);
    signed char* WvT    = (signed char*)alloc((size_t)KVDIM * DIM);
    signed char* WpT    = (signed char*)alloc((size_t)DIM * DIM);
    signed char* Xq     = (signed char*)alloc((size_t)ROWS * DIM);
    float*       gammaX = (float*)alloc((size_t)ROWS * sizeof(float));
    float*       Qp     = (float*)alloc((size_t)ROWS * DIM * 4);
    float*       Kp     = (float*)alloc((size_t)ROWS * KVDIM * 4);
    float*       Vp     = (float*)alloc((size_t)ROWS * KVDIM * 4);
    _Float16*    Qh     = (_Float16*)alloc((size_t)ROWS * DIM * 2);
    _Float16*    Kh     = (_Float16*)alloc((size_t)ROWS * KVDIM * 2);
    _Float16*    Vh     = (_Float16*)alloc((size_t)ROWS * KVDIM * 2);

    alpha_kernel<<<1, 256, 0, stream>>>(Wq, DIM * DIM,   alphas + 0);
    alpha_kernel<<<1, 256, 0, stream>>>(Wk, KVDIM * DIM, alphas + 1);
    alpha_kernel<<<1, 256, 0, stream>>>(Wv, KVDIM * DIM, alphas + 2);
    alpha_kernel<<<1, 256, 0, stream>>>(Wp, DIM * DIM,   alphas + 3);
    quant_w_kernel<<<(DIM * DIM) / 256,   256, 0, stream>>>(Wq, alphas + 0, WqT, DIM * DIM);
    quant_w_kernel<<<(KVDIM * DIM) / 256, 256, 0, stream>>>(Wk, alphas + 1, WkT, KVDIM * DIM);
    quant_w_kernel<<<(KVDIM * DIM) / 256, 256, 0, stream>>>(Wv, alphas + 2, WvT, KVDIM * DIM);
    quant_w_kernel<<<(DIM * DIM) / 256,   256, 0, stream>>>(Wp, alphas + 3, WpT, DIM * DIM);

    quant_rows_kernel<<<ROWS, 256, 0, stream>>>(x, Xq, gammaX, DIM);
    gemm_iu8_kernel<<<(ROWS / 16) * (DIM / 64) / 8,   256, 0, stream>>>(
        Xq, WqT, gammaX, alphas + 0, Qp, ROWS, DIM,   DIM);
    gemm_iu8_kernel<<<(ROWS / 16) * (KVDIM / 64) / 8, 256, 0, stream>>>(
        Xq, WkT, gammaX, alphas + 1, Kp, ROWS, KVDIM, DIM);
    gemm_iu8_kernel<<<(ROWS / 16) * (KVDIM / 64) / 8, 256, 0, stream>>>(
        Xq, WvT, gammaX, alphas + 2, Vp, ROWS, KVDIM, DIM);

    int qWaves = ROWS * NUM_HEADS, kWaves = ROWS * NUM_KVH;
    normrope_kernel<<<qWaves * 32 / 256, 256, 0, stream>>>(Qp, Qh, gain, NUM_HEADS, 2, 0.125f, qWaves);
    normrope_kernel<<<kWaves * 32 / 256, 256, 0, stream>>>(Kp, Kh, nullptr, NUM_KVH, 1, 1.0f, kWaves);
    normrope_kernel<<<kWaves * 32 / 256, 256, 0, stream>>>(Vp, Vh, nullptr, NUM_KVH, 0, 1.0f, kWaves);

    float* Y = Qp;
    attn_kernel<<<BSZ * NUM_HEADS * (SEQ / 128), 256, 0, stream>>>(Qh, Kh, Vh, Y);

    quant_rows_kernel<<<ROWS, 256, 0, stream>>>(Y, Xq, gammaX, DIM);
    gemm_iu8_kernel<<<(ROWS / 16) * (DIM / 64) / 8, 256, 0, stream>>>(
        Xq, WpT, gammaX, alphas + 3, out, ROWS, DIM, DIM);
}
